// LocalMultiheadAttention_15135464751248
// MI455X (gfx1250) — matrix operand order repaired and checked by reading, not compiled, not run on hardware
//
#include <hip/hip_runtime.h>

// ---------------------------------------------------------------------------
// Types / helpers for CDNA5 WMMA (wave32, v_wmma_f32_16x16x32_bf16)
// ---------------------------------------------------------------------------
typedef __attribute__((ext_vector_type(16))) __bf16 v16bf;
typedef __attribute__((ext_vector_type(8)))  float  v8f;

union ABfrag {
    v16bf        v;
    unsigned int u[8];
    uint4        q[2];
};

// Single-instruction packed f32->bf16 conversion (RNE under default MODE):
//   dst[15:0] = bf16(f0), dst[31:16] = bf16(f1)
__device__ __forceinline__ unsigned int bf16pair(float f0, float f1) {
    unsigned int r;
    asm("v_cvt_pk_bf16_f32 %0, %1, %2" : "=v"(r) : "v"(f0), "v"(f1));
    return r;
}
__device__ __forceinline__ unsigned short bf16of(float f) {
    return (unsigned short)(bf16pair(f, f) & 0xffffu);
}
__device__ __forceinline__ v8f wmma_bf16(const ABfrag& a, const ABfrag& b, v8f c) {
    // D = A(16x32 bf16) * B(32x16 bf16) + C(16x16 f32)
    return __builtin_amdgcn_wmma_f32_16x16x32_bf16(false, a.v, false, b.v,
                                                   (short)0, c, false, false);
}

// CDNA5 async global->LDS copy (tracked by ASYNCcnt), GVS addressing:
//   LDS[vdst_lds_off] = MEM[saddr + voffset]
__device__ __forceinline__ void async_ld128(unsigned lds_off, unsigned voff, const void* sbase) {
    asm volatile("global_load_async_to_lds_b128 %0, %1, %2"
                 :: "v"(lds_off), "v"(voff), "s"(sbase)
                 : "memory");
}
__device__ __forceinline__ void wait_async16() { asm volatile("s_wait_asynccnt 0x10" ::: "memory"); }
__device__ __forceinline__ void wait_async0()  { asm volatile("s_wait_asynccnt 0x0"  ::: "memory"); }

// Problem constants
constexpr int BATCH = 8;
constexpr int SEQ   = 4000;
constexpr int SEQP  = 4096;   // padded to multiple of WINDOW
constexpr int DIM   = 512;
constexpr int WIN   = 128;
constexpr int NW    = SEQP / WIN;   // 32 windows
constexpr int KW    = 3 * WIN;      // 384 keys per window (look back/fwd 1)

// ---------------------------------------------------------------------------
// Kernel 1: Y(bf16, padded) = (X @ W^T + b) * scale
// Software-pipelined: double-buffered LDS + register prefetch of next K-slice.
// Block: 256 threads = 8 waves; tile 128x128; each wave 32 rows x 64 cols.
// ---------------------------------------------------------------------------
__device__ __forceinline__ void proj_fetch(const float* __restrict__ X,
                                           const float* __restrict__ W,
                                           int batch, int row0, int col0,
                                           int lr, int lc, int kt,
                                           float4 fA[4], float4 fB[4]) {
    const int  grow = row0 + lr;
    const bool ok   = grow < SEQ;
    const int  crow = ok ? grow : (SEQ - 1);               // clamp, zero later
    const float* srcA = X + ((size_t)batch * SEQ + crow) * DIM + kt * 32 + lc;
    const float* srcB = W + (size_t)(col0 + lr) * DIM + kt * 32 + lc;
#pragma unroll
    for (int i = 0; i < 4; ++i) {
        float4 a = ((const float4*)srcA)[i];
        if (!ok) a = make_float4(0.f, 0.f, 0.f, 0.f);      // v_cndmask, no branch
        fA[i] = a;
        fB[i] = ((const float4*)srcB)[i];
    }
}

__global__ __launch_bounds__(256) void proj_gemm(const float* __restrict__ X,
                                                 const float* __restrict__ W,
                                                 const float* __restrict__ bias,
                                                 unsigned short* __restrict__ Y,
                                                 float scale) {
    __shared__ unsigned int As[2][128][16];   // 128 rows x 32 k as bf16 pairs
    __shared__ unsigned int Bs[2][128][16];   // 128 out-cols x 32 k as bf16 pairs

    const int t    = threadIdx.x;
    const int wv   = t >> 5;
    const int lane = t & 31;
    const int half = lane >> 4;
    const int ln   = lane & 15;

    const int mtile = blockIdx.x;
    const int batch = mtile >> 5;             // 32 row-tiles per batch
    const int row0  = (mtile & 31) * 128;
    const int col0  = blockIdx.y * 128;

    const int wrow = (wv & 3) * 32;
    const int wcol = (wv >> 2) * 64;

    v8f c[2][4];
#pragma unroll
    for (int i = 0; i < 2; ++i)
#pragma unroll
        for (int s = 0; s < 4; ++s) c[i][s] = (v8f){0.f,0.f,0.f,0.f,0.f,0.f,0.f,0.f};

    const int lr = t >> 1;
    const int lc = (t & 1) * 16;

    float4 fA[4], fB[4];
    proj_fetch(X, W, batch, row0, col0, lr, lc, 0, fA, fB);

    for (int kt = 0; kt < 16; ++kt) {
        const int sl = kt & 1;
        {   // pack current K-slice to LDS (one v_cvt_pk_bf16_f32 per pair)
            unsigned int* dA = &As[sl][lr][lc >> 1];
            unsigned int* dB = &Bs[sl][lr][lc >> 1];
#pragma unroll
            for (int i = 0; i < 4; ++i) {
                dA[2 * i + 0] = bf16pair(fA[i].x, fA[i].y);
                dA[2 * i + 1] = bf16pair(fA[i].z, fA[i].w);
                dB[2 * i + 0] = bf16pair(fB[i].x, fB[i].y);
                dB[2 * i + 1] = bf16pair(fB[i].z, fB[i].w);
            }
        }
        if (kt < 15)   // prefetch next K-slice; latency hides under compute
            proj_fetch(X, W, batch, row0, col0, lr, lc, kt + 1, fA, fB);
        __syncthreads();

        ABfrag a[2], b[4];
#pragma unroll
        for (int i = 0; i < 2; ++i) {
            const unsigned int* rA = As[sl][wrow + i * 16 + ln];
            a[i].q[0] = *(const uint4*)&rA[4 * half];       // K 0..7   (this half)
            a[i].q[1] = *(const uint4*)&rA[8 + 4 * half];   // K 16..23 (this half)
        }
#pragma unroll
        for (int s = 0; s < 4; ++s) {
            const unsigned int* rB = Bs[sl][wcol + s * 16 + ln];
            b[s].q[0] = *(const uint4*)&rB[4 * half];
            b[s].q[1] = *(const uint4*)&rB[8 + 4 * half];
        }
#pragma unroll
        for (int i = 0; i < 2; ++i)
#pragma unroll
            for (int s = 0; s < 4; ++s) c[i][s] = wmma_bf16(a[i], b[s], c[i][s]);
    }

    // Epilogue: bias, scale, pack to bf16, store to padded workspace
#pragma unroll
    for (int i = 0; i < 2; ++i)
#pragma unroll
        for (int s = 0; s < 4; ++s) {
            const int   gcol = col0 + wcol + s * 16 + ln;
            const float bv   = bias[gcol];
#pragma unroll
            for (int r = 0; r < 8; ++r) {
                const int grow = row0 + wrow + i * 16 + r + 8 * half;
                const float v  = (c[i][s][r] + bv) * scale;
                Y[((size_t)batch * SEQP + grow) * DIM + gcol] = bf16of(v);
            }
        }
}

// ---------------------------------------------------------------------------
// Kernel 2: windowed attention. One block per (batch, window).
// 256 threads = 8 waves; wave wv owns query rows [16*wv, 16*wv+16).
// Phase 1 staging uses double-buffered global_load_async_to_lds_b128.
// LDS map (bytes):
//   [0,1536)          valid flags (384 u32)
//   [1536,1664)       async dump line for masked-off lanes
//   [1664, +98304)    attn bf16 128x384
//   [99968, +2*65536) two stage slots: qs 128x64 bf16 (16384) + ks 384x64 bf16 (49152)
//   phase 2 reuses slot 0 as vT[64][392] bf16 (50176)
// ---------------------------------------------------------------------------
constexpr int OFF_DUMP  = 1536;
constexpr int OFF_ATTN  = 1664;
constexpr int LDS_ATTN  = 128 * KW * 2;                  // 98304
constexpr int OFF_STAGE = OFF_ATTN + LDS_ATTN;           // 99968
constexpr int SLOT_SZ   = 65536;                         // qs + ks
constexpr int OFF_KS    = 16384;                         // ks offset within a slot
constexpr int ATTN_LDS  = OFF_STAGE + 2 * SLOT_SZ;       // 231040 bytes (< 320 KB)
constexpr int VT_PITCH  = 392;                           // u16 pitch (bank-conflict pad)

__global__ __launch_bounds__(256) void local_attn(const unsigned short* __restrict__ Qb,
                                                  const unsigned short* __restrict__ Kb,
                                                  const unsigned short* __restrict__ Vb,
                                                  const unsigned char* __restrict__ mask,
                                                  float* __restrict__ Out) {
    extern __shared__ char smem[];
    const unsigned  smem0  = (unsigned)(unsigned long long)(void*)smem;  // LDS byte offset
    unsigned int*   validL = (unsigned int*)smem;
    unsigned short* attnS  = (unsigned short*)(smem + OFF_ATTN);
    char*           stage  = smem + OFF_STAGE;
    unsigned int*   qsP[2] = {(unsigned int*)stage, (unsigned int*)(stage + SLOT_SZ)};
    unsigned int*   ksP[2] = {(unsigned int*)(stage + OFF_KS),
                              (unsigned int*)(stage + SLOT_SZ + OFF_KS)};
    unsigned short* vt16   = (unsigned short*)stage;                // phase 2
    unsigned int*   vt32   = (unsigned int*)stage;

    const int t    = threadIdx.x;
    const int wv   = t >> 5;
    const int lane = t & 31;
    const int half = lane >> 4;
    const int ln   = lane & 15;

    const int b    = blockIdx.x >> 5;
    const int w    = blockIdx.x & 31;
    const int key0 = w * WIN - WIN;           // first key index of the 384-window

    // --- key validity: in [0,SEQ) and mask set ---
    if (t < KW) {
        const int idx = key0 + t;
        unsigned int v = 0;
        if (idx >= 0 && idx < SEQ) v = mask[(size_t)b * SEQ + idx] ? 1u : 0u;
        validL[t] = v;
    }

    const size_t qbase = ((size_t)b * SEQP + (size_t)w * WIN) * DIM;
    const size_t kvb   = (size_t)b * SEQP * DIM;
    const char*  qsrc  = (const char*)(Qb + qbase);
    const char*  ksrc  = (const char*)(Kb + kvb);

    // --- zero-fill OOB key rows in BOTH ks slots (async loads never touch them)
#pragma unroll
    for (int i = 0; i < 12; ++i) {
        const int li4 = i * 256 + t, r = li4 >> 3, c4 = li4 & 7;
        const int kidx = key0 + r;
        if (kidx < 0 || kidx >= SEQP) {
            const uint4 z = make_uint4(0u, 0u, 0u, 0u);
            *(uint4*)(ksP[0] + r * 32 + c4 * 4) = z;
            *(uint4*)(ksP[1] + r * 32 + c4 * 4) = z;
        }
    }

    // --- async stage issue: exactly 16 async ops per wave per stage ---
    auto issueStage = [&](int dc, int slot) {
        const unsigned qsL = smem0 + OFF_STAGE + slot * SLOT_SZ;
        const unsigned ksL = qsL + OFF_KS;
#pragma unroll
        for (int i = 0; i < 4; ++i) {            // q: 128x64 bf16
            const int li4 = i * 256 + t, r = li4 >> 3, c4 = li4 & 7;
            async_ld128(qsL + (unsigned)(r * 128 + c4 * 16),
                        (unsigned)(r * (DIM * 2) + dc * 128 + c4 * 16), qsrc);
        }
#pragma unroll
        for (int i = 0; i < 12; ++i) {           // k: 384x64 bf16
            const int li4 = i * 256 + t, r = li4 >> 3, c4 = li4 & 7;
            const int  kidx = key0 + r;
            const bool ok   = (kidx >= 0) && (kidx < SEQP);
            const unsigned lds  = ok ? ksL + (unsigned)(r * 128 + c4 * 16)
                                     : smem0 + OFF_DUMP;
            const unsigned voff = ok ? (unsigned)(kidx * (DIM * 2) + dc * 128 + c4 * 16) : 0u;
            async_ld128(lds, voff, ksrc);        // always issued: uniform ASYNCcnt
        }
    };

    __syncthreads();                             // validL + zero-fill visible
    unsigned int vbits = 0;                      // validity of col s*16+ln
#pragma unroll
    for (int s = 0; s < 24; ++s) vbits |= (validL[s * 16 + ln] << s);

    // --- phase 1: sim = q @ k^T, accumulated in registers (16x384 per wave) ---
    v8f c[24];
#pragma unroll
    for (int s = 0; s < 24; ++s) c[s] = (v8f){0.f,0.f,0.f,0.f,0.f,0.f,0.f,0.f};

    issueStage(0, 0);
    for (int dc = 0; dc < 8; ++dc) {             // 64-wide d chunks
        const int sl = dc & 1;
        __syncthreads();                          // slot (dc+1)&1 free for refill
        if (dc < 7) { issueStage(dc + 1, sl ^ 1); wait_async16(); }
        else        { wait_async0(); }
        __syncthreads();                          // all waves' stage dc in LDS
#pragma unroll
        for (int kk = 0; kk < 2; ++kk) {
            ABfrag a;
            const unsigned int* rQ = qsP[sl] + (wv * 16 + ln) * 32 + kk * 16;
            a.q[0] = *(const uint4*)&rQ[8 * half];
            a.q[1] = *(const uint4*)&rQ[8 * half + 4];
#pragma unroll
            for (int s = 0; s < 24; ++s) {
                ABfrag bf;
                const unsigned int* rK = ksP[sl] + (s * 16 + ln) * 32 + kk * 16 + 8 * half;
                bf.q[0] = *(const uint4*)&rK[0];
                bf.q[1] = *(const uint4*)&rK[4];
                c[s] = wmma_bf16(a, bf, c[s]);
            }
        }
    }

    // --- phase 1.5: masked softmax, fully in registers (rows live in 16-lane halves)
#pragma unroll
    for (int r = 0; r < 8; ++r) {
        float m = -3.0e38f;
#pragma unroll
        for (int s = 0; s < 24; ++s) {
            const float v = ((vbits >> s) & 1u) ? c[s][r] : -3.0e38f;
            m = fmaxf(m, v);
        }
        for (int off = 1; off < 16; off <<= 1) m = fmaxf(m, __shfl_xor(m, off, 32));
        float sum = 0.f;
#pragma unroll
        for (int s = 0; s < 24; ++s) {
            const float v = ((vbits >> s) & 1u) ? __expf(c[s][r] - m) : 0.f;
            c[s][r] = v;
            sum += v;
        }
        for (int off = 1; off < 16; off <<= 1) sum += __shfl_xor(sum, off, 32);
        const float is   = (sum > 0.f) ? 1.f / sum : 0.f;
        const int   mrow = wv * 16 + r + 8 * half;
#pragma unroll
        for (int s = 0; s < 24; ++s)
            attnS[mrow * KW + s * 16 + ln] = bf16of(c[s][r] * is);
    }
    __syncthreads();   // attnS complete; stage region now reusable

    // --- phase 2: out = attn @ v, streaming 64-wide d chunks of v (transposed)
    const unsigned int* attn32 = (const unsigned int*)attnS;   // pitch 192 u32
    for (int dc = 0; dc < 8; ++dc) {
        __syncthreads();  // all reads of previous vT done
#pragma unroll
        for (int i = 0; i < 12; ++i) {
            const int li4 = i * 256 + t, r = li4 >> 3, c4 = li4 & 7;
            const int kidx = key0 + r;
            uint4 d = make_uint4(0u, 0u, 0u, 0u);
            if (kidx >= 0 && kidx < SEQP)
                d = *(const uint4*)(Vb + kvb + (size_t)kidx * DIM + dc * 64 + c4 * 8);
            const unsigned int uu[4] = {d.x, d.y, d.z, d.w};
#pragma unroll
            for (int e = 0; e < 4; ++e) {
                const int dd = c4 * 8 + e * 2;
                vt16[(dd + 0) * VT_PITCH + r] = (unsigned short)(uu[e] & 0xffffu);
                vt16[(dd + 1) * VT_PITCH + r] = (unsigned short)(uu[e] >> 16);
            }
        }
        __syncthreads();

        v8f o[4];
#pragma unroll
        for (int s = 0; s < 4; ++s) o[s] = (v8f){0.f,0.f,0.f,0.f,0.f,0.f,0.f,0.f};
#pragma unroll
        for (int kk = 0; kk < 12; ++kk) {                 // K = 384 keys
            ABfrag a;
            const unsigned int* rA = attn32 + (wv * 16 + ln) * (KW / 2) + kk * 16;
            a.q[0] = *(const uint4*)&rA[8 * half];
            a.q[1] = *(const uint4*)&rA[8 * half + 4];
#pragma unroll
            for (int s = 0; s < 4; ++s) {
                const int dd = s * 16 + ln;
                ABfrag bf;
                const unsigned int* rV = vt32 + dd * (VT_PITCH / 2) + kk * 16 + 8 * half;
                bf.q[0] = *(const uint4*)&rV[0];
                bf.q[1] = *(const uint4*)&rV[4];
                o[s] = wmma_bf16(a, bf, o[s]);
            }
        }
        // write fp32 output, cropping padded query rows
#pragma unroll
        for (int s = 0; s < 4; ++s) {
#pragma unroll
            for (int r = 0; r < 8; ++r) {
                const int qrow = w * WIN + wv * 16 + r + 8 * half;
                if (qrow < SEQ)
                    Out[((size_t)b * SEQ + qrow) * DIM + dc * 64 + s * 16 + ln] = o[s][r];
            }
        }
    }
}

// ---------------------------------------------------------------------------
// Launch
// ---------------------------------------------------------------------------
extern "C" void kernel_launch(void* const* d_in, const int* in_sizes, int n_in,
                              void* d_out, int out_size, void* d_ws, size_t ws_size,
                              hipStream_t stream) {
    const float*         queries = (const float*)d_in[0];
    const float*         keys    = (const float*)d_in[1];
    const float*         values  = (const float*)d_in[2];
    const unsigned char* maskp   = (const unsigned char*)d_in[3];
    const float*         w_q     = (const float*)d_in[4];
    const float*         b_q     = (const float*)d_in[5];
    const float*         w_k     = (const float*)d_in[6];
    const float*         b_k     = (const float*)d_in[7];
    const float*         w_v     = (const float*)d_in[8];
    const float*         b_v     = (const float*)d_in[9];

    const size_t projElems = (size_t)BATCH * SEQP * DIM;   // bf16 elements each
    unsigned short* Qb = (unsigned short*)d_ws;
    unsigned short* Kb = Qb + projElems;
    unsigned short* Vb = Kb + projElems;

    const dim3 gp(BATCH * SEQP / 128, DIM / 128);          // (256, 4)
    const float SCALE = 0.125f;                            // 64^-0.5
    proj_gemm<<<gp, 256, 0, stream>>>(queries, w_q, b_q, Qb, SCALE);
    proj_gemm<<<gp, 256, 0, stream>>>(keys,    w_k, b_k, Kb, 1.0f);
    proj_gemm<<<gp, 256, 0, stream>>>(values,  w_v, b_v, Vb, 1.0f);

    local_attn<<<BATCH * NW, 256, ATTN_LDS, stream>>>(Qb, Kb, Vb, maskp, (float*)d_out);
}
